// ScaledDotAttentionNAIM_28750511079452
// MI455X (gfx1250) — matrix operand order repaired and checked by reading, not compiled, not run on hardware
//
#include <hip/hip_runtime.h>
#include <hip/hip_bf16.h>
#include <math.h>

// CDNA5 WMMA vector types (wave32): 16 bf16 per lane = 16x32 bf16 matrix per wave
typedef __attribute__((ext_vector_type(16))) __bf16 v16bf;
typedef __attribute__((ext_vector_type(8)))  float  v8f;

// async-to-LDS builtin takes int4* in AS1 (global) / AS3 (LDS)
typedef int v4i __attribute__((vector_size(16)));
typedef __attribute__((address_space(1))) v4i as1_v4i;
typedef __attribute__((address_space(3))) v4i as3_v4i;

#if __has_builtin(__builtin_amdgcn_global_load_async_to_lds_b128)
#define HAVE_ASYNC_LDS 1
#else
#define HAVE_ASYNC_LDS 0
#endif

#if __has_builtin(__builtin_amdgcn_s_wait_asynccnt)
#define WAIT_ASYNC0() __builtin_amdgcn_s_wait_asynccnt(0)
#else
#define WAIT_ASYNC0() asm volatile("s_wait_asynccnt 0x0" ::: "memory")
#endif

#define B_  2
#define L_  2048
#define S_  2048
#define H_  16
#define E_  64
#define TM  16          // query rows per workgroup
#define NWAVES 8
#define SCALE 0.125f    // 1/sqrt(64)
#define EPS_  1e-8f

// Dynamic LDS layout (floats):
//   sc[TM][S_]           32768
//   red[16][16]            256
//   rowmax[16]              16
//   rowinv[16]              16
//   pacc[4][256]          1024   (phase-3 partial C combine)
//   vstage[8][32*16]      4096   (phase-3 per-wave V staging)
#define SMEM_FLOATS (TM * S_ + 256 + 16 + 16 + 1024 + 4096)

__global__ __launch_bounds__(256)
void attn_naim_kernel(const float* __restrict__ q,
                      const float* __restrict__ k,
                      const float* __restrict__ v,
                      const unsigned char* __restrict__ mk,   // mask_miss_k (bool, 1B)
                      const unsigned char* __restrict__ mq,   // mask_miss_q (bool, 1B)
                      const int* __restrict__ pos,            // (B,L,1) int32
                      const int* __restrict__ causal,         // scalar int
                      float* __restrict__ outV,
                      float* __restrict__ outA,
                      float* __restrict__ outE)
{
    extern __shared__ float smem[];
    float* sc     = smem;                 // TM * S_ scores / probs
    float* red    = sc + TM * S_;         // 16 x 16 reduction scratch
    float* rowmax = red + 256;            // 16
    float* rowinv = rowmax + 16;          // 16
    float* pacc   = rowinv + 16;          // 4 * 256
    float* vstage = pacc + 1024;          // 8 * 512 (16B aligned)

    const int bid = blockIdx.x;
    const int lt  = bid % (L_ / TM);
    const int h   = (bid / (L_ / TM)) % H_;
    const int b   = bid / ((L_ / TM) * H_);
    const int l0  = lt * TM;

    const int tid  = threadIdx.x;
    const int wave = tid >> 5;
    const int lane = tid & 31;
    const int lrow = lane & 15;     // 0..15 : matrix row (A) / column (B,D)
    const int hi   = lane >> 4;     // half-wave selector per ISA layouts
    const bool do_causal = (causal[0] != 0);

    // ---------------- Phase 1: scores = scale * (Q K^T) with masks -> LDS ----
    // A-matrix (Q) fragment, 16x32, contiguous k order chosen here (same as B):
    //   lane m (both halves) holds row M=m; VGPR i: K = hi*16 + i*2
    v16bf a0, a1;
    {
        const float* qrow = q + (((size_t)b * L_ + (l0 + lrow)) * H_ + h) * E_;
        #pragma unroll
        for (int c = 0; c < 2; ++c) {
            v16bf& a = c ? a1 : a0;
            #pragma unroll
            for (int i = 0; i < 8; ++i) {
                int e = c * 32 + hi * 16 + i * 2;
                float2 f = *(const float2*)(qrow + e);
                a[2 * i]     = (__bf16)f.x;
                a[2 * i + 1] = (__bf16)f.y;
            }
        }
    }
    // pos of the 8 rows this lane produces in the D fragment
    int rpos[8];
    #pragma unroll
    for (int r = 0; r < 8; ++r)
        rpos[r] = pos[(size_t)b * L_ + l0 + hi * 8 + r];

    for (int t = wave; t < S_ / 16; t += NWAVES) {
        const int s0   = t * 16;
        const int scol = s0 + lrow;
        // B-matrix (K^T) fragment: lane n -> column s0+n; lanes 0-15 hold
        // K-dim e=0..15 (of the 32-chunk), lanes 16-31 hold e=16..31.
        v16bf b0, b1;
        const float* krow = k + (((size_t)b * S_ + scol) * H_ + h) * E_;
        #pragma unroll
        for (int c = 0; c < 2; ++c) {
            v16bf& bb = c ? b1 : b0;
            const int ebase = c * 32 + hi * 16;
            #pragma unroll
            for (int j = 0; j < 16; j += 4) {
                float4 f = *(const float4*)(krow + ebase + j);
                bb[j]     = (__bf16)f.x;
                bb[j + 1] = (__bf16)f.y;
                bb[j + 2] = (__bf16)f.z;
                bb[j + 3] = (__bf16)f.w;
            }
        }
        v8f c8 = {};
        c8 = __builtin_amdgcn_wmma_f32_16x16x32_bf16(false, a0, false, b0,
                                                     (short)0, c8, false, false);
        c8 = __builtin_amdgcn_wmma_f32_16x16x32_bf16(false, a1, false, b1,
                                                     (short)0, c8, false, false);

        const bool kmiss = (mk[(size_t)b * S_ + scol] != 0);
        const int  pcol  = pos[(size_t)b * L_ + scol];
        // D layout: VGPR r, lanes 0-15 -> (M=r, N=lane), lanes 16-31 -> (M=r+8)
        #pragma unroll
        for (int r = 0; r < 8; ++r) {
            const int m = r + hi * 8;
            float x = c8[r] * SCALE;
            const bool masked = kmiss || (do_causal && (pcol > rpos[r]));
            sc[m * S_ + scol] = masked ? -INFINITY : x;
        }
    }
    __syncthreads();

    // ---------------- Phase 2: softmax + masks + A writeback + entropy -------
    const int row = tid >> 4;     // 0..15
    const int sub = tid & 15;     // 0..15, strided columns
    const size_t outAbase = (((size_t)b * H_ + h) * L_ + l0) * S_;

    {   // row max
        float m = -INFINITY;
        for (int c = sub; c < S_; c += 16) m = fmaxf(m, sc[row * S_ + c]);
        red[row * 16 + sub] = m;
    }
    __syncthreads();
    if (sub == 0) {
        float m = -INFINITY;
        #pragma unroll
        for (int i = 0; i < 16; ++i) m = fmaxf(m, red[row * 16 + i]);
        rowmax[row] = m;
    }
    __syncthreads();
    {   // exp + partial sum (all -inf row -> zeros, mirroring nan_to_num)
        const float rm = rowmax[row];
        float s = 0.f;
        for (int c = sub; c < S_; c += 16) {
            const float x = sc[row * S_ + c];
            const float e = (rm == -INFINITY || x == -INFINITY) ? 0.f : __expf(x - rm);
            sc[row * S_ + c] = e;
            s += e;
        }
        red[row * 16 + sub] = s;
    }
    __syncthreads();
    if (sub == 0) {
        float s = 0.f;
        #pragma unroll
        for (int i = 0; i < 16; ++i) s += red[row * 16 + i];
        const bool qmiss = (mq[(size_t)b * L_ + l0 + row] != 0);
        // missing query: att = max(softmax - inf, 0) = 0 -> inv = 0
        rowinv[row] = (qmiss || !(s > 0.f)) ? 0.f : 1.f / s;
    }
    __syncthreads();
    {   // normalize, stream A to HBM (non-temporal: 536 MB write-once stream),
        // entropy partials, keep probs in LDS for phase 3
        const float inv = rowinv[row];
        float ent = 0.f;
        for (int c = sub; c < S_; c += 16) {
            const float a = sc[row * S_ + c] * inv;
            sc[row * S_ + c] = a;
            __builtin_nontemporal_store(a, &outA[outAbase + (size_t)row * S_ + c]);
            ent -= a * __logf(fmaxf(a, EPS_));     // a==0 contributes exactly 0
        }
        red[row * 16 + sub] = ent;
    }
    __syncthreads();
    if (sub == 0) {
        float e = 0.f;
        #pragma unroll
        for (int i = 0; i < 16; ++i) e += red[row * 16 + i];
        outE[((size_t)b * H_ + h) * L_ + l0 + row] = e;
    }
    __syncthreads();

    // ---------------- Phase 3: V_out = A (16 x S) * V (S x E), bf16 WMMA -----
    // All 8 waves active: waves w and w+4 share E-tile (w&3), each takes half
    // the S/K range; partials combined through LDS.
    {
        const int e0 = (wave & 3) * 16;
        const int kh = wave >> 2;                 // K-half: 0 or 1
        float* wbase = vstage + wave * 512;       // 32 rows x 16 cols staging
        v8f acc = {};
        for (int kc = kh * 32; kc < kh * 32 + 32; ++kc) {
#if HAVE_ASYNC_LDS
            // Stage V rows [kc*32, kc*32+32), cols [e0, e0+16) into LDS with
            // coalesced async copies (4 x b128 per lane-task, ASYNCcnt tracked).
            {
                #pragma unroll
                for (int i = 0; i < 4; ++i) {
                    const int task = i * 32 + lane;          // 0..127
                    const int r  = task >> 2;                // V row within chunk
                    const int qq = task & 3;                 // 16B quarter of row
                    const float* gp = v + (((size_t)b * S_ + kc * 32 + r) * H_ + h) * E_
                                        + e0 + qq * 4;
                    float* lp = wbase + r * 16 + qq * 4;
                    __builtin_amdgcn_global_load_async_to_lds_b128(
                        (as1_v4i*)gp, (as3_v4i*)lp, 0, 0);
                }
                WAIT_ASYNC0();
            }
#endif
            // A fragment from LDS probabilities (f32 -> bf16)
            v16bf af;
            #pragma unroll
            for (int i = 0; i < 8; ++i) {
                const int so = kc * 32 + hi * 16 + i * 2;
                af[2 * i]     = (__bf16)sc[lrow * S_ + so];
                af[2 * i + 1] = (__bf16)sc[lrow * S_ + so + 1];
            }
            // B fragment: lane n -> column e0+n; element j -> local row hi*16+j
            v16bf bf_;
#if HAVE_ASYNC_LDS
            #pragma unroll
            for (int j = 0; j < 16; ++j)
                bf_[j] = (__bf16)wbase[(hi * 16 + j) * 16 + lrow];
#else
            const float* vcol = v + (((size_t)b * S_ + kc * 32 + hi * 16) * H_ + h) * E_
                                  + e0 + lrow;
            #pragma unroll
            for (int j = 0; j < 16; ++j)
                bf_[j] = (__bf16)vcol[(size_t)j * (H_ * E_)];
#endif
            acc = __builtin_amdgcn_wmma_f32_16x16x32_bf16(false, af, false, bf_,
                                                          (short)0, acc, false, false);
        }
        // combine the two K-halves via LDS, then store
        if (wave >= 4) {
            #pragma unroll
            for (int r = 0; r < 8; ++r)
                pacc[(wave & 3) * 256 + r * 32 + lane] = acc[r];
        }
        __syncthreads();
        if (wave < 4) {
            #pragma unroll
            for (int r = 0; r < 8; ++r) {
                const int m = r + hi * 8;
                const float val = acc[r] + pacc[wave * 256 + r * 32 + lane];
                outV[(((size_t)b * L_ + l0 + m) * H_ + h) * E_ + e0 + lrow] = val;
            }
        }
    }
}

extern "C" void kernel_launch(void* const* d_in, const int* in_sizes, int n_in,
                              void* d_out, int out_size, void* d_ws, size_t ws_size,
                              hipStream_t stream) {
    (void)in_sizes; (void)n_in; (void)d_ws; (void)ws_size; (void)out_size;
    const float*         q      = (const float*)d_in[0];
    const float*         k      = (const float*)d_in[1];
    const float*         v      = (const float*)d_in[2];
    const unsigned char* mk     = (const unsigned char*)d_in[3]; // bool arrays: 1B/elem
    const unsigned char* mq     = (const unsigned char*)d_in[4];
    const int*           pos    = (const int*)d_in[5];
    const int*           causal = (const int*)d_in[6];

    float* out  = (float*)d_out;
    float* outV = out;                                        // (B,L,H,E)
    float* outA = outV + (size_t)B_ * L_ * H_ * E_;           // (B,H,L,S)
    float* outE = outA + (size_t)B_ * H_ * L_ * S_;           // (B,H,L)

    const size_t smem = (size_t)SMEM_FLOATS * sizeof(float);  // ~149 KB: 2 WGs / WGP
    (void)hipFuncSetAttribute(reinterpret_cast<const void*>(attn_naim_kernel),
                              hipFuncAttributeMaxDynamicSharedMemorySize, (int)smem);

    dim3 grid(B_ * H_ * (L_ / TM));   // 4096 workgroups
    attn_naim_kernel<<<grid, 256, smem, stream>>>(q, k, v, mk, mq, pos, causal,
                                                  outV, outA, outE);
}
